// NeighborEmbedding_14628658610722
// MI455X (gfx1250) — matrix operand order repaired and checked by reading, not compiled, not run on hardware
//
#include <hip/hip_runtime.h>
#include <hip/hip_bf16.h>

#define EPSV 1e-5f

typedef __attribute__((ext_vector_type(16))) _Float16 v16h;
typedef __attribute__((ext_vector_type(8)))  _Float16 v8h;
typedef __attribute__((ext_vector_type(8)))  float    v8f;

// ---------------- WMMA helpers (CDNA5 16x16x32 f16 -> f32) ----------------
// A fragment per lane: row M = lane&15. halves 0..7 = K kk+8g .. +7,
// halves 8..15 = K kk+16+8g .. +7   (CDNA5 ISA 16-bit A layout)
static __device__ __forceinline__ v16h frag_a(const _Float16* row, int kk, int g) {
  const v8h lo = *(const v8h*)(row + kk + 8 * g);
  const v8h hi = *(const v8h*)(row + kk + 16 + 8 * g);
  v16h r;
#pragma unroll
  for (int i = 0; i < 8; ++i) { r[i] = lo[i]; r[i + 8] = hi[i]; }
  return r;
}

// B fragment per lane: col N = n (row of weight matrix, since B = W^T),
// K kk+8g .. +7 then kk+16+8g .. +7 (same as A). Weights row-major (o,d) f16.
static __device__ __forceinline__ v16h frag_b(const _Float16* w, int Din, int n,
                                              int kk, int g) {
  const _Float16* p = w + (size_t)n * Din + kk + 8 * g;
  const v8h lo = *(const v8h*)(p);
  const v8h hi = *(const v8h*)(p + 16);
  v16h r;
#pragma unroll
  for (int i = 0; i < 8; ++i) { r[i] = lo[i]; r[i + 8] = hi[i]; }
  return r;
}

static __device__ __forceinline__ v8f wmma16(v16h a, v16h b, v8f c) {
  return __builtin_amdgcn_wmma_f32_16x16x32_f16(false, a, false, b, (short)0, c,
                                                false, false);
}

// Async global->LDS 16B copy (gfx1250): no VGPR round-trip, ASYNCcnt tracked.
// LDS offset = low 32 bits of generic pointer (aperture in high half).
static __device__ __forceinline__ void async_copy_b128(void* lds_dst, const void* gsrc) {
  unsigned lds_off = (unsigned)(size_t)lds_dst;
  asm volatile("global_load_async_to_lds_b128 %0, %1, off"
               :: "v"(lds_off), "v"(gsrc) : "memory");
}
static __device__ __forceinline__ void wait_async0() {
  asm volatile("s_wait_asynccnt 0" ::: "memory");
}

// ---------------- small prep kernels ----------------
__global__ void prep_bn_kernel(const float* __restrict__ g, const float* __restrict__ b,
                               const float* __restrict__ m, const float* __restrict__ v,
                               float* __restrict__ sb, int C) {
  int i = blockIdx.x * blockDim.x + threadIdx.x;
  if (i < C) {
    float s = g[i] * rsqrtf(v[i] + EPSV);
    sb[i] = s;
    sb[C + i] = b[i] - m[i] * s;
  }
}

__global__ void cvt_f16_kernel(const float* __restrict__ src, _Float16* __restrict__ dst, int n) {
  int i = blockIdx.x * blockDim.x + threadIdx.x;
  if (i < n) dst[i] = (_Float16)src[i];
}

__global__ void transpose_coords_kernel(const float* __restrict__ x, float* __restrict__ coords,
                                        int B, int N) {
  int i = blockIdx.x * blockDim.x + threadIdx.x;
  if (i < B * N) {
    int b = i / N, n = i - b * N;
    const float* xb = x + (size_t)b * 3 * N;
    coords[(size_t)i * 3 + 0] = xb[0 * N + n];
    coords[(size_t)i * 3 + 1] = xb[1 * N + n];
    coords[(size_t)i * 3 + 2] = xb[2 * N + n];
  }
}

// ---------------- embed: f = bnrelu(bnrelu(x W1^T) W2^T), conv2 via WMMA ----
__global__ void __launch_bounds__(256)
embed_kernel(const float* __restrict__ x, const float* __restrict__ w1,
             const float* __restrict__ bn1sb, const _Float16* __restrict__ w2h,
             const float* __restrict__ bn2sb, float* __restrict__ f0, int N) {
  extern __shared__ char smem[];
  _Float16* h1 = (_Float16*)smem;  // 128 x 64 f16
  int chunks = N / 128;
  int b = blockIdx.x / chunks;
  int p0 = (blockIdx.x - b * chunks) * 128;
  int t = threadIdx.x;

  // stage 1: conv1 (K=3, scalar) + bn1 + relu -> LDS f16
  {
    int pt = t >> 1;
    int half = t & 1;
    int n = p0 + pt;
    const float* xb = x + (size_t)b * 3 * N;
    float x0 = xb[0 * N + n], x1 = xb[1 * N + n], x2 = xb[2 * N + n];
#pragma unroll
    for (int j = 0; j < 32; ++j) {
      int o = half * 32 + j;
      float h = x0 * w1[o * 3 + 0] + x1 * w1[o * 3 + 1] + x2 * w1[o * 3 + 2];
      h = fmaxf(h * bn1sb[o] + bn1sb[64 + o], 0.f);
      h1[pt * 64 + o] = (_Float16)h;
    }
  }
  __syncthreads();

  // stage 2: conv2 (64x64) via WMMA, each wave = 16 points x 64 channels
  int wave = t >> 5, lane = t & 31, g = lane >> 4, l15 = lane & 15;
  int m0 = wave * 16;
  const _Float16* arow = h1 + (m0 + l15) * 64;
#pragma unroll
  for (int nt = 0; nt < 4; ++nt) {
    int n = nt * 16 + l15;
    v8f c = {};
#pragma unroll
    for (int kk = 0; kk < 64; kk += 32) {
      v16h a = frag_a(arow, kk, g);
      v16h bf = frag_b(w2h, 64, n, kk, g);
      c = wmma16(a, bf, c);
    }
    float sc = bn2sb[n], bi = bn2sb[64 + n];
#pragma unroll
    for (int r = 0; r < 8; ++r) {
      int pt = m0 + r + 8 * g;
      float v = fmaxf(c[r] * sc + bi, 0.f);
      f0[((size_t)b * N + p0 + pt) * 64 + n] = v;
    }
  }
}

// ---------------- farthest point sampling (one workgroup per batch) --------
// Serial S-step loop; per step: wave32 shfl argmax + one 32-entry LDS stage.
__global__ void fps_kernel(const float* __restrict__ coords, int N, int S,
                           int* __restrict__ idx) {
  extern __shared__ char smem[];
  float* dist = (float*)smem;            // N
  float* rv = dist + N;                  // 32
  int* ri = (int*)(rv + 32);             // 32
  __shared__ int sFar;
  int b = blockIdx.x, t = threadIdx.x, T = blockDim.x;
  int lane = t & 31, wave = t >> 5, nw = T >> 5;
  const float* cb = coords + (size_t)b * N * 3;
  for (int j = t; j < N; j += T) dist[j] = 1e10f;
  if (t == 0) sFar = 0;
  __syncthreads();
  for (int s = 0; s < S; ++s) {
    int far = sFar;
    if (t == 0) idx[b * S + s] = far;
    float cx = cb[far * 3 + 0], cy = cb[far * 3 + 1], cz = cb[far * 3 + 2];
    float lm = -1.f; int li = 0x7fffffff;
    for (int j = t; j < N; j += T) {
      float dx = cb[j * 3 + 0] - cx, dy = cb[j * 3 + 1] - cy, dz = cb[j * 3 + 2] - cz;
      float dd = fminf(dist[j], dx * dx + dy * dy + dz * dz);
      dist[j] = dd;
      if (dd > lm) { lm = dd; li = j; }  // ascending scan keeps lowest index
    }
    // wave32 argmax (ties -> lower index)
#pragma unroll
    for (int off = 16; off > 0; off >>= 1) {
      float ov = __shfl_xor(lm, off, 32);
      int oi = __shfl_xor(li, off, 32);
      if (ov > lm || (ov == lm && oi < li)) { lm = ov; li = oi; }
    }
    if (lane == 0) { rv[wave] = lm; ri[wave] = li; }
    __syncthreads();
    if (wave == 0) {
      float v = (lane < nw) ? rv[lane] : -1.f;
      int vi = (lane < nw) ? ri[lane] : 0x7fffffff;
#pragma unroll
      for (int off = 16; off > 0; off >>= 1) {
        float ov = __shfl_xor(v, off, 32);
        int oi = __shfl_xor(vi, off, 32);
        if (ov > v || (ov == v && oi < vi)) { v = ov; vi = oi; }
      }
      if (lane == 0) sFar = vi;
    }
    __syncthreads();
  }
}

__global__ void gather_xyz_kernel(const float* __restrict__ coords, const int* __restrict__ idx,
                                  float* __restrict__ out, int N, int S, int total) {
  int i = blockIdx.x * blockDim.x + threadIdx.x;
  if (i < total) {
    int b = i / S;
    int p = idx[i];
    const float* src = coords + ((size_t)b * N + p) * 3;
    out[(size_t)i * 3 + 0] = src[0];
    out[(size_t)i * 3 + 1] = src[1];
    out[(size_t)i * 3 + 2] = src[2];
  }
}

// ---------------- kNN: one wave32 per query, iterative argmin selection ----
__global__ void knn_kernel(const float* __restrict__ coords, const float* __restrict__ query,
                           int N, int S, int Kn, int* __restrict__ nidx) {
  extern __shared__ float sdist[];  // N
  int q = blockIdx.x;               // b*S + s
  int b = q / S;
  int lane = threadIdx.x;
  const float* cb = coords + (size_t)b * N * 3;
  float qx = query[(size_t)q * 3 + 0], qy = query[(size_t)q * 3 + 1], qz = query[(size_t)q * 3 + 2];
  float q2 = qx * qx + qy * qy + qz * qz;
  float lm = 1e30f; int li = 0x7fffffff;
  for (int j = lane; j < N; j += 32) {
    float cx = cb[j * 3 + 0], cy = cb[j * 3 + 1], cz = cb[j * 3 + 2];
    float d = q2 - 2.f * (qx * cx + qy * cy + qz * cz) + (cx * cx + cy * cy + cz * cz);
    sdist[j] = d;
    if (d < lm) { lm = d; li = j; }
  }
  for (int k = 0; k < Kn; ++k) {
    float v = lm; int vi = li;
#pragma unroll
    for (int off = 16; off > 0; off >>= 1) {
      float ov = __shfl_xor(v, off, 32);
      int oi = __shfl_xor(vi, off, 32);
      if (ov < v || (ov == v && oi < vi)) { v = ov; vi = oi; }
    }
    if (lane == 0) nidx[(size_t)q * Kn + k] = vi;
    if ((vi & 31) == lane) {        // owner lane removes & rescans its slice
      sdist[vi] = 1e30f;
      lm = 1e30f; li = 0x7fffffff;
      for (int j = lane; j < N; j += 32) {
        float d = sdist[j];
        if (d < lm) { lm = d; li = j; }
      }
    }
  }
}

// ---------------- set-abstraction MLP block (WMMA + async gather) ----------
// One workgroup (8 wave32s) per (b,s) group of 32 neighbors.
// async-gather rows -> agg(32 x 2C f16) -> GEMM1 -> bn+relu -> h1(32 x D f16)
// -> GEMM2 -> bn+relu -> max over neighbors.
__global__ void __launch_bounds__(256)
sg_mlp_kernel(const float* __restrict__ feats, int featRows,
              const int* __restrict__ fpsIdx, const int* __restrict__ nidx,
              const _Float16* __restrict__ w1h, const _Float16* __restrict__ w2h,
              const float* __restrict__ bn1sb, const float* __restrict__ bn2sb,
              float* __restrict__ outp, int S, int C, int D, int transposed) {
  const int Din = 2 * C;
  extern __shared__ char smem[];
  float* stage = (float*)smem;                       // 33 x C f32 (row 32 = center)
  _Float16* agg = (_Float16*)(stage + 33 * C);       // 32 x Din
  _Float16* h1 = agg + 32 * Din;                     // 32 x D
  int gidx = blockIdx.x;                             // b*S + s
  int b = gidx / S, s = gidx - b * S;
  int t = threadIdx.x;
  const float* fb = feats + (size_t)b * featRows * C;
  int pidx = fpsIdx[gidx];
  const int* ni = nidx + (size_t)gidx * 32;

  // async global->LDS gather of 33 feature rows, 16B chunks per lane
  {
    int cpr = C >> 2;                 // 16B chunks per row
    int total = 33 * cpr;
    for (int e = t; e < total; e += 256) {
      int k = e / cpr, c16 = e - k * cpr;
      int row = (k < 32) ? ni[k] : pidx;
      async_copy_b128(stage + k * C + c16 * 4, fb + (size_t)row * C + c16 * 4);
    }
  }
  wait_async0();
  __syncthreads();

  // build agg = [grouped - center, center] in f16
  const float* ctr = stage + 32 * C;
  for (int e = t; e < 32 * Din; e += 256) {
    int k = e / Din, d = e - k * Din;
    float val = (d < C) ? (stage[k * C + d] - ctr[d]) : ctr[d - C];
    agg[k * Din + d] = (_Float16)val;
  }
  __syncthreads();

  int wave = t >> 5, lane = t & 31, g = lane >> 4, l15 = lane & 15;
  int ntPerWave = D >> 7;  // D/16 tiles over 8 waves

  // GEMM1: h1 = relu(bn1(agg @ w1^T))
  for (int nti = 0; nti < ntPerWave; ++nti) {
    int n = (wave * ntPerWave + nti) * 16 + l15;
    float sc = bn1sb[n], bi = bn1sb[D + n];
    for (int mt = 0; mt < 2; ++mt) {
      const _Float16* arow = agg + (mt * 16 + l15) * Din;
      v8f c = {};
      for (int kk = 0; kk < Din; kk += 32) {
        __builtin_prefetch(w1h + (size_t)n * Din + kk + 64, 0, 1);
        v16h a = frag_a(arow, kk, g);
        v16h bf = frag_b(w1h, Din, n, kk, g);
        c = wmma16(a, bf, c);
      }
#pragma unroll
      for (int r = 0; r < 8; ++r) {
        int row = mt * 16 + r + 8 * g;
        h1[row * D + n] = (_Float16)fmaxf(c[r] * sc + bi, 0.f);
      }
    }
  }
  __syncthreads();

  // GEMM2 + bn2 + relu + max over the 32 neighbor rows
  for (int nti = 0; nti < ntPerWave; ++nti) {
    int n = (wave * ntPerWave + nti) * 16 + l15;
    float sc = bn2sb[n], bi = bn2sb[D + n];
    float mx = 0.f;  // relu outputs are >= 0
    for (int mt = 0; mt < 2; ++mt) {
      const _Float16* arow = h1 + (mt * 16 + l15) * D;
      v8f c = {};
      for (int kk = 0; kk < D; kk += 32) {
        __builtin_prefetch(w2h + (size_t)n * D + kk + 64, 0, 1);
        v16h a = frag_a(arow, kk, g);
        v16h bf = frag_b(w2h, D, n, kk, g);
        c = wmma16(a, bf, c);
      }
#pragma unroll
      for (int r = 0; r < 8; ++r) mx = fmaxf(mx, fmaxf(c[r] * sc + bi, 0.f));
    }
    mx = fmaxf(mx, __shfl_xor(mx, 16, 32));  // combine M 0-7 with 8-15
    if (lane < 16) {
      if (transposed) outp[((size_t)b * D + n) * S + s] = mx;       // (B, D, S)
      else            outp[((size_t)(b * S + s)) * D + n] = mx;     // (B, S, D)
    }
  }
}

// ---------------- host orchestration ----------------
extern "C" void kernel_launch(void* const* d_in, const int* in_sizes, int n_in,
                              void* d_out, int out_size, void* d_ws, size_t ws_size,
                              hipStream_t stream) {
  (void)in_sizes; (void)n_in; (void)out_size; (void)ws_size;
  const int B = 32, N = 8192, S0 = 256, S1 = 128, Kn = 32;

  const float* x       = (const float*)d_in[0];
  const float* conv1_w = (const float*)d_in[1];
  const float* bn1p[4] = {(const float*)d_in[2], (const float*)d_in[3], (const float*)d_in[4], (const float*)d_in[5]};
  const float* conv2_w = (const float*)d_in[6];
  const float* bn2p[4] = {(const float*)d_in[7], (const float*)d_in[8], (const float*)d_in[9], (const float*)d_in[10]};
  const float* s0w1    = (const float*)d_in[11];
  const float* s0b1[4] = {(const float*)d_in[12], (const float*)d_in[13], (const float*)d_in[14], (const float*)d_in[15]};
  const float* s0w2    = (const float*)d_in[16];
  const float* s0b2[4] = {(const float*)d_in[17], (const float*)d_in[18], (const float*)d_in[19], (const float*)d_in[20]};
  const float* s1w1    = (const float*)d_in[21];
  const float* s1b1[4] = {(const float*)d_in[22], (const float*)d_in[23], (const float*)d_in[24], (const float*)d_in[25]};
  const float* s1w2    = (const float*)d_in[26];
  const float* s1b2[4] = {(const float*)d_in[27], (const float*)d_in[28], (const float*)d_in[29], (const float*)d_in[30]};

  char* p = (char*)d_ws;
  auto carve = [&](size_t bytes) -> char* {
    char* r = p; p += (bytes + 255) & ~(size_t)255; return r;
  };
  float*    coords = (float*)carve((size_t)B * N * 3 * 4);
  float*    f0     = (float*)carve((size_t)B * N * 64 * 4);
  float*    f1     = (float*)carve((size_t)B * S0 * 128 * 4);
  float*    xyz1   = (float*)carve((size_t)B * S0 * 3 * 4);
  float*    xyz2   = (float*)carve((size_t)B * S1 * 3 * 4);
  int*      idx0   = (int*)carve((size_t)B * S0 * 4);
  int*      idx1   = (int*)carve((size_t)B * S1 * 4);
  int*      nidx0  = (int*)carve((size_t)B * S0 * Kn * 4);
  int*      nidx1  = (int*)carve((size_t)B * S1 * Kn * 4);
  _Float16* w2h    = (_Float16*)carve((size_t)64 * 64 * 2);
  _Float16* s0w1h  = (_Float16*)carve((size_t)128 * 128 * 2);
  _Float16* s0w2h  = (_Float16*)carve((size_t)128 * 128 * 2);
  _Float16* s1w1h  = (_Float16*)carve((size_t)256 * 256 * 2);
  _Float16* s1w2h  = (_Float16*)carve((size_t)256 * 256 * 2);
  float*    bn1sb  = (float*)carve(64 * 2 * 4);
  float*    bn2sb  = (float*)carve(64 * 2 * 4);
  float*    s0b1sb = (float*)carve(128 * 2 * 4);
  float*    s0b2sb = (float*)carve(128 * 2 * 4);
  float*    s1b1sb = (float*)carve(256 * 2 * 4);
  float*    s1b2sb = (float*)carve(256 * 2 * 4);

  // BN folding + weight f16 conversion
  prep_bn_kernel<<<1, 64, 0, stream>>>(bn1p[0], bn1p[1], bn1p[2], bn1p[3], bn1sb, 64);
  prep_bn_kernel<<<1, 64, 0, stream>>>(bn2p[0], bn2p[1], bn2p[2], bn2p[3], bn2sb, 64);
  prep_bn_kernel<<<1, 128, 0, stream>>>(s0b1[0], s0b1[1], s0b1[2], s0b1[3], s0b1sb, 128);
  prep_bn_kernel<<<1, 128, 0, stream>>>(s0b2[0], s0b2[1], s0b2[2], s0b2[3], s0b2sb, 128);
  prep_bn_kernel<<<1, 256, 0, stream>>>(s1b1[0], s1b1[1], s1b1[2], s1b1[3], s1b1sb, 256);
  prep_bn_kernel<<<1, 256, 0, stream>>>(s1b2[0], s1b2[1], s1b2[2], s1b2[3], s1b2sb, 256);
  cvt_f16_kernel<<<(64 * 64 + 255) / 256, 256, 0, stream>>>(conv2_w, w2h, 64 * 64);
  cvt_f16_kernel<<<(128 * 128 + 255) / 256, 256, 0, stream>>>(s0w1, s0w1h, 128 * 128);
  cvt_f16_kernel<<<(128 * 128 + 255) / 256, 256, 0, stream>>>(s0w2, s0w2h, 128 * 128);
  cvt_f16_kernel<<<(256 * 256 + 255) / 256, 256, 0, stream>>>(s1w1, s1w1h, 256 * 256);
  cvt_f16_kernel<<<(256 * 256 + 255) / 256, 256, 0, stream>>>(s1w2, s1w2h, 256 * 256);

  // coords (B,N,3) and point embedding f0 (B,N,64)
  transpose_coords_kernel<<<(B * N + 255) / 256, 256, 0, stream>>>(x, coords, B, N);
  embed_kernel<<<B * (N / 128), 256, 128 * 64 * 2, stream>>>(x, conv1_w, bn1sb, w2h, bn2sb, f0, N);

  // stage 0: FPS -> gather xyz -> kNN -> MLP block (WMMA) -> f1 (B,S0,128)
  fps_kernel<<<B, 1024, N * 4 + 32 * 8, stream>>>(coords, N, S0, idx0);
  gather_xyz_kernel<<<(B * S0 + 255) / 256, 256, 0, stream>>>(coords, idx0, xyz1, N, S0, B * S0);
  knn_kernel<<<B * S0, 32, N * 4, stream>>>(coords, xyz1, N, S0, Kn, nidx0);
  {
    size_t lds = (size_t)33 * 64 * 4 + (size_t)32 * 128 * 2 + (size_t)32 * 128 * 2;
    sg_mlp_kernel<<<B * S0, 256, lds, stream>>>(f0, N, idx0, nidx0, s0w1h, s0w2h,
                                                s0b1sb, s0b2sb, f1, S0, 64, 128, 0);
  }

  // stage 1: FPS over 256 pts -> kNN -> MLP block -> d_out transposed (B,256,128)
  fps_kernel<<<B, 256, S0 * 4 + 32 * 8, stream>>>(xyz1, S0, S1, idx1);
  gather_xyz_kernel<<<(B * S1 + 255) / 256, 256, 0, stream>>>(xyz1, idx1, xyz2, S0, S1, B * S1);
  knn_kernel<<<B * S1, 32, S0 * 4, stream>>>(xyz1, xyz2, S0, S1, Kn, nidx1);
  {
    size_t lds = (size_t)33 * 128 * 4 + (size_t)32 * 256 * 2 + (size_t)32 * 256 * 2;
    sg_mlp_kernel<<<B * S1, 256, lds, stream>>>(f1, S0, idx1, nidx1, s1w1h, s1w2h,
                                                s1b1sb, s1b2sb, (float*)d_out, S1, 128, 256, 1);
  }
}
